// MMAttention_60344290509083
// MI455X (gfx1250) — matrix operand order repaired and checked by reading, not compiled, not run on hardware
//
#include <hip/hip_runtime.h>
#include <hip/hip_bf16.h>
#include <stdint.h>

typedef __bf16 bf16_t;
typedef __attribute__((ext_vector_type(16))) __bf16 v16bf;
typedef __attribute__((ext_vector_type(8)))  __bf16 v8bf;
typedef __attribute__((ext_vector_type(8)))  float  v8f;

#define BATCH 2
#define TSEQ  2048
#define EMB   1024
#define NH    16
#define HD    64
#define NREL  4095   // 2*TSEQ-1

union FragU { v16bf v; v8bf h[2]; };

static __device__ inline v16bf load_afrag(const bf16_t* p) {
  // A-fragment: elements 0..15 contiguous at p (k = 16*half + s)
  FragU u;
  u.h[0] = *(const v8bf*)(p);
  u.h[1] = *(const v8bf*)(p + 8);
  return u.v;
}

static __device__ inline v8f wmma_bf16(v16bf a, v16bf b, v8f c) {
  return __builtin_amdgcn_wmma_f32_16x16x32_bf16(false, a, false, b, (short)0, c,
                                                 false, false);
}

// ---------------- fp32 -> bf16 conversion ----------------
__global__ void cvt_f32_bf16(const float* __restrict__ in, bf16_t* __restrict__ out, int n) {
  int i = blockIdx.x * blockDim.x + threadIdx.x;
  int stride = gridDim.x * blockDim.x;
  for (; i < n; i += stride) out[i] = (bf16_t)in[i];
}

// ---------------- per-token GRU gate: gate[(b*NH+h)*T + t] ----------------
__global__ void gate_kernel(const float* __restrict__ hs, const float* __restrict__ gw,
                            const float* __restrict__ gb, const float* __restrict__ gc,
                            float* __restrict__ gate) {
  int idx = blockIdx.x * blockDim.x + threadIdx.x;   // (b*NH + h)*TSEQ + t
  if (idx >= BATCH * NH * TSEQ) return;
  int t = idx & (TSEQ - 1);
  int h = (idx >> 11) & (NH - 1);
  int b = idx >> 15;
  const float* x = hs + ((size_t)(b * TSEQ + t)) * EMB + h * HD;
  float a[8];
#pragma unroll
  for (int j = 0; j < 8; ++j) a[j] = gb[j];
  for (int d = 0; d < HD; ++d) {
    float xv = x[d];
#pragma unroll
    for (int j = 0; j < 8; ++j) a[j] += xv * gw[j * HD + d];
  }
  float pa = a[0] + a[1] + a[2] + a[3];
  float pb = a[4] + a[5] + a[6] + a[7];
  float ga = 1.0f / (1.0f + __expf(-pa));
  float gbv = 1.0f / (1.0f + __expf(-pb));
  gate[idx] = ga * (gbv * gc[h] - 1.0f) + 2.0f;
}

// ---------------- relative position bias table: rb[h*NREL + (rel+2047)] ----------------
__global__ void relbias_kernel(const float* __restrict__ rel_embed, float* __restrict__ rb) {
  int idx = blockIdx.x * blockDim.x + threadIdx.x;   // r*NH + h
  if (idx >= NREL * NH) return;
  int h = idx & (NH - 1);
  int r = idx >> 4;
  int rp = r - (TSEQ - 1);           // mem - ctx = s - t
  int bucket = (rp > 0) ? 160 : 0;
  int rel = rp < 0 ? -rp : rp;
  if (rel < 80) {
    bucket += rel;
  } else {
    float large = __logf((float)rel * (1.0f / 80.0f)) * (80.0f / __logf(10.0f));
    int li = 80 + (int)large;
    if (li > 159) li = 159;
    bucket += li;
  }
  rb[h * NREL + r] = rel_embed[bucket * NH + h];
}

// ---------------- WMMA GEMM: C = A(M,1024) * W(N,1024)^T (+bias)*scale ----------------
// one wave computes a 32x64 tile (2 A-frags reused against 4 B-frags -> 8 wmma/step).
// modes: 0=q (B,H,T,D bf16), 1=k (B,H,T,D bf16), 2=v transposed (B,H,D,T bf16), 3=fp32
__global__ __launch_bounds__(32)
void gemm_wmma(const bf16_t* __restrict__ A, const bf16_t* __restrict__ W,
               const float* __restrict__ bias, void* __restrict__ out,
               int mode, float scale) {
  const int K = EMB;
  int lane = threadIdx.x;
  int hlf  = lane >> 4;
  int m0 = blockIdx.x * 32;
  int n0 = blockIdx.y * 64;

  const bf16_t* ap0 = A + (size_t)(m0 + (lane & 15)) * K + (hlf << 4);
  const bf16_t* ap1 = ap0 + (size_t)16 * K;
  int ncol = lane & 15;
  int kbB  = hlf << 4;

  v8f acc[2][4] = {};
  for (int k0 = 0; k0 < K; k0 += 32) {
    v16bf a0 = load_afrag(ap0 + k0);
    v16bf a1 = load_afrag(ap1 + k0);
    if (k0 + 256 < K) {                       // stream-ahead (global_prefetch_b8)
      __builtin_prefetch(ap0 + k0 + 256, 0, 1);
      __builtin_prefetch(ap1 + k0 + 256, 0, 1);
    }
#pragma unroll
    for (int j = 0; j < 4; ++j) {
      const bf16_t* bp = W + (size_t)(n0 + j * 16 + ncol) * K + k0 + kbB;
      v16bf b = *(const v16bf*)bp;
      acc[0][j] = wmma_bf16(a0, b, acc[0][j]);
      acc[1][j] = wmma_bf16(a1, b, acc[1][j]);
    }
  }

#pragma unroll
  for (int mi = 0; mi < 2; ++mi) {
#pragma unroll
    for (int j = 0; j < 4; ++j) {
      int n = n0 + j * 16 + ncol;
      float bn = bias ? bias[n] : 0.0f;
      if (mode == 3) {
        float* O = (float*)out;
#pragma unroll
        for (int r = 0; r < 8; ++r) {
          int m = m0 + mi * 16 + r + 8 * hlf;
          O[(size_t)m * EMB + n] = (acc[mi][j][r] + bn) * scale;
        }
      } else {
        bf16_t* O = (bf16_t*)out;
        int h = n >> 6, d = n & (HD - 1);
#pragma unroll
        for (int r = 0; r < 8; ++r) {
          int m = m0 + mi * 16 + r + 8 * hlf;
          int b = m >> 11, t = m & (TSEQ - 1);
          float v = (acc[mi][j][r] + bn) * scale;
          size_t idx;
          if (mode == 2) idx = (((size_t)(b * NH + h) * HD + d) * TSEQ + t);
          else           idx = (((size_t)(b * NH + h) * TSEQ + t) * HD + d);
          O[idx] = (bf16_t)v;
        }
      }
    }
  }
}

// ---------------- flash attention with gated relative bias ----------------
// grid: (T/16, B*NH); one wave per block handles 16 queries of one (b,h).
// K/V tiles for each 32-key block are double-buffered in LDS via the gfx1250
// async-to-LDS engine (ASYNCcnt), overlapping the next block's 6KB stream-in
// with the current block's wmma + online softmax.
__global__ __launch_bounds__(32)
void attn_wmma(const bf16_t* __restrict__ qT, const bf16_t* __restrict__ kT,
               const bf16_t* __restrict__ vT2, const float* __restrict__ gate,
               const float* __restrict__ relbias, bf16_t* __restrict__ ctx) {
  __shared__ __align__(32) bf16_t kbuf[2][32 * HD];  // 2 x 4KB (contiguous rows of kT)
  __shared__ __align__(32) bf16_t vbuf[2][HD * 32];  // 2 x 4KB (rows of vT2)
  __shared__ __align__(32) bf16_t pbuf[16 * 32];     // P re-layout tile
  int lane = threadIdx.x;
  int hlf  = lane >> 4;
  int t0 = blockIdx.x * 16;
  int bh = blockIdx.y;              // b*NH + h
  int h = bh & (NH - 1);
  int b = bh >> 4;
  int scol = lane & 15;

  // issue one 32-key block's K (first 2KB only, keys 0..15) + V (64 rows x 64B) = 12 async b128/lane
  auto issue_tile = [&](int s0n, int buf) {
    uint32_t kl = (uint32_t)(uintptr_t)(&kbuf[buf][0]) + (uint32_t)(lane * 16);
    uint64_t kg = (uint64_t)(uintptr_t)(kT + ((size_t)bh * TSEQ + s0n) * HD) +
                  (uint64_t)(lane * 16);
#pragma unroll
    for (int p = 0; p < 4; ++p) {
      uint32_t la = kl + p * 512;
      uint64_t ga = kg + p * 512;
      asm volatile("global_load_async_to_lds_b128 %0, %1, off"
                   :: "v"(la), "v"(ga) : "memory");
    }
#pragma unroll
    for (int p = 0; p < 2; ++p) {
      int d = p * 32 + lane;
      uint32_t vl = (uint32_t)(uintptr_t)(&vbuf[buf][0]) + (uint32_t)(d * 64);
      uint64_t vg = (uint64_t)(uintptr_t)(vT2 + ((size_t)bh * HD + d) * TSEQ + s0n);
      asm volatile("global_load_async_to_lds_b128 %0, %1, off"
                   :: "v"(vl), "v"(vg) : "memory");
      asm volatile("global_load_async_to_lds_b128 %0, %1, off offset:16"
                   :: "v"(vl), "v"(vg) : "memory");
      asm volatile("global_load_async_to_lds_b128 %0, %1, off offset:32"
                   :: "v"(vl), "v"(vg) : "memory");
      asm volatile("global_load_async_to_lds_b128 %0, %1, off offset:48"
                   :: "v"(vl), "v"(vg) : "memory");
    }
  };

  // Q fragments (16 rows x 64 K), pre-scaled by D^-0.5 at projection time
  const bf16_t* qbase = qT + ((size_t)bh * TSEQ + t0 + scol) * HD + (hlf << 4);
  v16bf qa0 = load_afrag(qbase);
  v16bf qa1 = load_afrag(qbase + 32);

  float mrow[8], lrow[8], grow[8];
  const float* gp = gate + (size_t)bh * TSEQ + t0;
#pragma unroll
  for (int r = 0; r < 8; ++r) {
    mrow[r] = -1e30f;
    lrow[r] = 0.0f;
    grow[r] = gp[r + 8 * hlf];
  }
  const float* rb = relbias + h * NREL + (TSEQ - 1);

  v8f o[4] = {};
  issue_tile(0, 0);
  int cur = 0;

  for (int s0 = 0; s0 < TSEQ; s0 += 32) {
    // prefetch next block, then wait for current buffer (async loads complete in order)
    if (s0 + 32 < TSEQ) {
      issue_tile(s0 + 32, cur ^ 1);
      asm volatile("s_wait_asynccnt 0xc" ::: "memory");
    } else {
      asm volatile("s_wait_asynccnt 0x0" ::: "memory");
    }

    // ---- scores: two 16x16 tiles over this 32-wide key block (K from LDS) ----
    v8f S[2];
#pragma unroll
    for (int u = 0; u < 2; ++u) {
      const bf16_t* kp = &kbuf[cur][(u * 16 + scol) * HD + (hlf << 4)];
      v16bf kb0 = *(const v16bf*)(kp);
      v16bf kb1 = *(const v16bf*)(kp + 32);
      v8f c = {};
      c = wmma_bf16(qa0, kb0, c);
      c = wmma_bf16(qa1, kb1, c);
      S[u] = c;
    }
    // ---- bias + online softmax (row reductions across 16 lanes) ----
#pragma unroll
    for (int r = 0; r < 8; ++r) {
      int t = t0 + r + 8 * hlf;
      float x0 = S[0][r] + grow[r] * rb[(s0 + scol) - t];
      float x1 = S[1][r] + grow[r] * rb[(s0 + 16 + scol) - t];
      float mx = fmaxf(x0, x1);
      mx = fmaxf(mx, __shfl_xor(mx, 1, 32));
      mx = fmaxf(mx, __shfl_xor(mx, 2, 32));
      mx = fmaxf(mx, __shfl_xor(mx, 4, 32));
      mx = fmaxf(mx, __shfl_xor(mx, 8, 32));
      float mnew = fmaxf(mrow[r], mx);
      float sc = __expf(mrow[r] - mnew);
      float p0 = __expf(x0 - mnew);
      float p1 = __expf(x1 - mnew);
      float rs = p0 + p1;
      rs += __shfl_xor(rs, 1, 32);
      rs += __shfl_xor(rs, 2, 32);
      rs += __shfl_xor(rs, 4, 32);
      rs += __shfl_xor(rs, 8, 32);
      lrow[r] = lrow[r] * sc + rs;
      mrow[r] = mnew;
#pragma unroll
      for (int j = 0; j < 4; ++j) o[j][r] *= sc;
      int rowIdx = r + 8 * hlf;
      pbuf[rowIdx * 32 + scol]      = (bf16_t)p0;
      pbuf[rowIdx * 32 + 16 + scol] = (bf16_t)p1;
    }
    // ---- re-layout P through LDS into an A-fragment (same-wave DS is in-order) ----
    v16bf pa = load_afrag(&pbuf[scol * 32 + (hlf << 4)]);
    // ---- O += P * V  (V B-fragments from LDS, contiguous rows of vT2) ----
#pragma unroll
    for (int j = 0; j < 4; ++j) {
      const bf16_t* vp = &vbuf[cur][(j * 16 + scol) * 32 + (hlf << 4)];
      v16bf vb = *(const v16bf*)vp;
      o[j] = wmma_bf16(pa, vb, o[j]);
    }
    cur ^= 1;
  }

  // ---- epilogue: normalize, store ctx (B,T,E) bf16 ----
#pragma unroll
  for (int j = 0; j < 4; ++j) {
    int e = h * HD + j * 16 + scol;
#pragma unroll
    for (int r = 0; r < 8; ++r) {
      int t = t0 + r + 8 * hlf;
      float val = o[j][r] / lrow[r];
      ctx[((size_t)(b * TSEQ + t)) * EMB + e] = (bf16_t)val;
    }
  }
}

extern "C" void kernel_launch(void* const* d_in, const int* in_sizes, int n_in,
                              void* d_out, int out_size, void* d_ws, size_t ws_size,
                              hipStream_t stream) {
  const float* hs  = (const float*)d_in[0];
  const float* q_w = (const float*)d_in[1];
  const float* q_b = (const float*)d_in[2];
  const float* k_w = (const float*)d_in[3];
  const float* k_b = (const float*)d_in[4];
  const float* v_w = (const float*)d_in[5];
  const float* v_b = (const float*)d_in[6];
  const float* o_w = (const float*)d_in[7];
  const float* o_b = (const float*)d_in[8];
  const float* rel = (const float*)d_in[9];
  const float* gc  = (const float*)d_in[10];
  const float* gw  = (const float*)d_in[11];
  const float* gb  = (const float*)d_in[12];

  char* w = (char*)d_ws;
  auto take = [&](size_t bytes) -> char* {
    char* p = w;
    w += (bytes + 255) & ~(size_t)255;
    return p;
  };
  const size_t actB = (size_t)BATCH * TSEQ * EMB;   // 4.19M elems
  const size_t wB   = (size_t)EMB * EMB;            // 1.05M elems
  bf16_t* hsb  = (bf16_t*)take(actB * 2);
  bf16_t* wq   = (bf16_t*)take(wB * 2);
  bf16_t* wk   = (bf16_t*)take(wB * 2);
  bf16_t* wv   = (bf16_t*)take(wB * 2);
  bf16_t* wo   = (bf16_t*)take(wB * 2);
  bf16_t* qT   = (bf16_t*)take(actB * 2);
  bf16_t* kT   = (bf16_t*)take(actB * 2);
  bf16_t* vT2  = (bf16_t*)take(actB * 2);
  bf16_t* ctxb = (bf16_t*)take(actB * 2);
  float*  gatep = (float*)take((size_t)BATCH * NH * TSEQ * 4);
  float*  rbp   = (float*)take((size_t)NH * NREL * 4);

  // precision conversions
  cvt_f32_bf16<<<4096, 256, 0, stream>>>(hs, hsb, (int)actB);
  cvt_f32_bf16<<<1024, 256, 0, stream>>>(q_w, wq, (int)wB);
  cvt_f32_bf16<<<1024, 256, 0, stream>>>(k_w, wk, (int)wB);
  cvt_f32_bf16<<<1024, 256, 0, stream>>>(v_w, wv, (int)wB);
  cvt_f32_bf16<<<1024, 256, 0, stream>>>(o_w, wo, (int)wB);

  // gate + relative-bias table
  gate_kernel<<<(BATCH * NH * TSEQ + 255) / 256, 256, 0, stream>>>(hs, gw, gb, gc, gatep);
  relbias_kernel<<<(NREL * NH + 255) / 256, 256, 0, stream>>>(rel, rbp);

  // QKV projections (q pre-scaled by D^-0.5; v stored transposed)
  dim3 gg(BATCH * TSEQ / 32, EMB / 64);
  gemm_wmma<<<gg, 32, 0, stream>>>(hsb, wq, q_b, qT, 0, 0.125f);
  gemm_wmma<<<gg, 32, 0, stream>>>(hsb, wk, k_b, kT, 1, 1.0f);
  gemm_wmma<<<gg, 32, 0, stream>>>(hsb, wv, v_b, vT2, 2, 1.0f);

  // flash attention with gated relative position bias
  attn_wmma<<<dim3(TSEQ / 16, BATCH * NH), 32, 0, stream>>>(qT, kT, vT2, gatep, rbp, ctxb);

  // output projection -> fp32 d_out
  gemm_wmma<<<gg, 32, 0, stream>>>(ctxb, wo, o_b, (float*)d_out, 3, 1.0f);
}
